// TransformerBlock_86526411145375
// MI455X (gfx1250) — matrix operand order repaired and checked by reading, not compiled, not run on hardware
//
#include <hip/hip_runtime.h>
#include <hip/hip_bf16.h>
#include <math.h>

typedef __attribute__((ext_vector_type(16))) _Float16     v16h;
typedef __attribute__((ext_vector_type(8)))  float        v8f;
typedef __attribute__((ext_vector_type(2)))  _Float16     h2;
typedef __attribute__((ext_vector_type(8)))  unsigned int v8u;

#define BATCH 4
#define DIM   192
#define HEADS 8
#define DH    24
#define HID   510
#define HIDP  512    // padded K for final GEMM
#define HID2  1020
#define HID2P 1024   // padded M for w_in GEMM
#define C3    576
#define HH    128
#define WW    128
#define PIX   16384
#define EPS_LN   1e-6f
#define EPS_NORM 1e-12f

__device__ __forceinline__ float gelu_exact(float x) {
  return 0.5f * x * (1.0f + erff(x * 0.70710678118654752f));
}

// ---------------- f32 -> f16 weight conversion with row/col zero padding ----------------
__global__ void k_cvt_pad(const float* __restrict__ s, _Float16* __restrict__ d,
                          int rows, int cols, int rowsPad, int colsPad) {
  int i = blockIdx.x * blockDim.x + threadIdx.x;
  if (i >= rowsPad * colsPad) return;
  int r = i / colsPad, c = i % colsPad;
  d[i] = (r < rows && c < cols) ? (_Float16)s[r * cols + c] : (_Float16)0.f;
}

// ---------------- channel LayerNorm (over C), output f16 ----------------
__global__ void k_chan_ln(const float* __restrict__ src, const float* __restrict__ gamma,
                          const float* __restrict__ beta, _Float16* __restrict__ dst, int C) {
  int idx = blockIdx.x * blockDim.x + threadIdx.x;   // b*PIX + p
  if (idx >= BATCH * PIX) return;
  int b = idx / PIX, p = idx % PIX;
  const float* base = src + (size_t)b * C * PIX + p;
  float s = 0.f, s2 = 0.f;
  for (int c = 0; c < C; ++c) {
    float v = base[(size_t)c * PIX];
    s += v; s2 += v * v;
  }
  float mean = s / (float)C;
  float var  = s2 / (float)C - mean * mean;
  float rstd = rsqrtf(var + EPS_LN);
  _Float16* ob = dst + (size_t)b * C * PIX + p;
  for (int c = 0; c < C; ++c) {
    float v = base[(size_t)c * PIX];
    ob[(size_t)c * PIX] = (_Float16)(((v - mean) * rstd) * gamma[c] + beta[c]);
  }
}

// ---------------- depthwise 3x3 conv, f16 in/out ----------------
__global__ void k_dwconv3_f16(const _Float16* __restrict__ src, const float* __restrict__ wdw,
                              _Float16* __restrict__ dst, int C) {
  size_t idx = (size_t)blockIdx.x * blockDim.x + threadIdx.x;
  size_t total = (size_t)BATCH * C * PIX;
  if (idx >= total) return;
  int p = (int)(idx % PIX);
  size_t bc = idx / PIX;
  int c = (int)(bc % C);
  int y = p >> 7, x = p & 127;
  const _Float16* sp = src + bc * PIX;
  const float* w = wdw + c * 9;
  float acc = 0.f;
  #pragma unroll
  for (int dy = -1; dy <= 1; ++dy) {
    int yy = y + dy;
    if (yy < 0 || yy >= HH) continue;
    #pragma unroll
    for (int dx = -1; dx <= 1; ++dx) {
      int xx = x + dx;
      if (xx < 0 || xx >= WW) continue;
      acc += (float)sp[yy * WW + xx] * w[(dy + 1) * 3 + (dx + 1)];
    }
  }
  dst[idx] = (_Float16)acc;
}

// ------- fused depthwise 3x3 + GEGLU gate: [B][1024-stride] 1020ch -> [B][512-stride] 510ch -------
__global__ void k_dwconv_geglu(const _Float16* __restrict__ src, const float* __restrict__ wdw,
                               _Float16* __restrict__ dst) {
  size_t idx = (size_t)blockIdx.x * blockDim.x + threadIdx.x;
  if (idx >= (size_t)BATCH * HID * PIX) return;
  int p = (int)(idx % PIX);
  size_t bi = idx / PIX;
  int i = (int)(bi % HID);
  int b = (int)(bi / HID);
  int y = p >> 7, x = p & 127;
  const _Float16* s1 = src + ((size_t)b * HID2P + i) * PIX;
  const _Float16* s2 = s1 + (size_t)HID * PIX;
  const float* w1 = wdw + i * 9;
  const float* w2 = wdw + (i + HID) * 9;
  float a1 = 0.f, a2 = 0.f;
  #pragma unroll
  for (int dy = -1; dy <= 1; ++dy) {
    int yy = y + dy;
    if (yy < 0 || yy >= HH) continue;
    #pragma unroll
    for (int dx = -1; dx <= 1; ++dx) {
      int xx = x + dx;
      if (xx < 0 || xx >= WW) continue;
      int o = yy * WW + xx, ki = (dy + 1) * 3 + (dx + 1);
      a1 += (float)s1[o] * w1[ki];
      a2 += (float)s2[o] * w2[ki];
    }
  }
  dst[((size_t)b * HIDP + i) * PIX + p] = (_Float16)(gelu_exact(a1) * a2);
}

// ---------------- gelu(dwconv3(x, w_local)) added into d_out (f32) ----------------
__global__ void k_local_ft_add(const float* __restrict__ x, const float* __restrict__ wdw,
                               float* __restrict__ out) {
  size_t idx = (size_t)blockIdx.x * blockDim.x + threadIdx.x;
  if (idx >= (size_t)BATCH * DIM * PIX) return;
  int p = (int)(idx % PIX);
  size_t bc = idx / PIX;
  int c = (int)(bc % DIM);
  int y = p >> 7, xx0 = p & 127;
  const float* sp = x + bc * PIX;
  const float* w = wdw + c * 9;
  float acc = 0.f;
  #pragma unroll
  for (int dy = -1; dy <= 1; ++dy) {
    int yy = y + dy;
    if (yy < 0 || yy >= HH) continue;
    #pragma unroll
    for (int dx = -1; dx <= 1; ++dx) {
      int xx = xx0 + dx;
      if (xx < 0 || xx >= WW) continue;
      acc += sp[yy * WW + xx] * w[(dy + 1) * 3 + (dx + 1)];
    }
  }
  out[idx] += gelu_exact(acc);
}

// ---------------- channel attention: Gram matrix + row norms + softmax ----------------
// qkv layout: [B][576][PIX] f16, q=ch[0..191], k=ch[192..383], v=ch[384..575]
__global__ void __launch_bounds__(576) k_attn_scores(const _Float16* __restrict__ qkv,
                                                     float* __restrict__ attn) {
  __shared__ float qs[DH * 65];
  __shared__ float ks[DH * 65];
  __shared__ float S[DH * DH];
  __shared__ float qn[DH];
  __shared__ float kn[DH];
  int bh = blockIdx.x;
  int b = bh / HEADS, hh = bh % HEADS;
  int t = threadIdx.x;                 // 0..575
  int d = t / DH, e = t % DH;
  const _Float16* qb = qkv + ((size_t)b * C3 + hh * DH) * PIX;
  const _Float16* kb = qb + (size_t)DIM * PIX;
  float acc = 0.f, nrm = 0.f;
  for (int n0 = 0; n0 < PIX; n0 += 64) {
    for (int i = t; i < DH * 64; i += 576) {
      int dd = i >> 6, j = i & 63;
      qs[dd * 65 + j] = (float)qb[(size_t)dd * PIX + n0 + j];
      ks[dd * 65 + j] = (float)kb[(size_t)dd * PIX + n0 + j];
    }
    __syncthreads();
    #pragma unroll 8
    for (int j = 0; j < 64; ++j) acc += qs[d * 65 + j] * ks[e * 65 + j];
    if (t < DH) {
      #pragma unroll 8
      for (int j = 0; j < 64; ++j) { float v = qs[t * 65 + j]; nrm += v * v; }
    } else if (t < 2 * DH) {
      int r = t - DH;
      #pragma unroll 8
      for (int j = 0; j < 64; ++j) { float v = ks[r * 65 + j]; nrm += v * v; }
    }
    __syncthreads();
  }
  S[t] = acc;
  if (t < DH) qn[t] = sqrtf(nrm);
  else if (t < 2 * DH) kn[t - DH] = sqrtf(nrm);
  __syncthreads();
  if (t < DH) {
    const float scale = rsqrtf((float)DH);
    float qd = fmaxf(qn[t], EPS_NORM);
    float row[DH];
    float m = -1e30f;
    #pragma unroll
    for (int ee = 0; ee < DH; ++ee) {
      float v = S[t * DH + ee] / (qd * fmaxf(kn[ee], EPS_NORM)) * scale;
      row[ee] = v;
      m = fmaxf(m, v);
    }
    float ssum = 0.f;
    #pragma unroll
    for (int ee = 0; ee < DH; ++ee) { row[ee] = expf(row[ee] - m); ssum += row[ee]; }
    float inv = 1.f / ssum;
    #pragma unroll
    for (int ee = 0; ee < DH; ++ee)
      attn[((size_t)bh * DH + t) * DH + ee] = row[ee] * inv;
  }
}

// ---------------- out[d,n] = sum_e attn[d,e] * v[e,n]  (f16 out, pre-1x1) ----------------
__global__ void k_attn_apply(const _Float16* __restrict__ qkv, const float* __restrict__ attn,
                             _Float16* __restrict__ outp) {
  __shared__ float A[DH * DH];
  int bh = blockIdx.y;
  int b = bh / HEADS, hh = bh % HEADS;
  int t = threadIdx.x;
  for (int i = t; i < DH * DH; i += blockDim.x) A[i] = attn[(size_t)bh * DH * DH + i];
  __syncthreads();
  int p = blockIdx.x * blockDim.x + t;
  const _Float16* vb = qkv + ((size_t)b * C3 + 2 * DIM + hh * DH) * PIX + p;
  float vv[DH];
  #pragma unroll
  for (int e = 0; e < DH; ++e) vv[e] = (float)vb[(size_t)e * PIX];
  _Float16* ob = outp + ((size_t)b * DIM + hh * DH) * PIX + p;
  #pragma unroll
  for (int d = 0; d < DH; ++d) {
    float acc = 0.f;
    #pragma unroll
    for (int e = 0; e < DH; ++e) acc += A[d * DH + e] * vv[e];
    ob[(size_t)d * PIX] = (_Float16)acc;
  }
}

// ---------------- WMMA GEMM: D[b][m][p] = sum_k W[m][k]*X[b][k][p] (+res) ----------------
// K, M padded to 32/16 multiples; NO runtime guards anywhere.
// Block: 256 thr = 8 waves; block owns one 16-row m-tile (A staged in LDS, shared);
// each wave computes a 16(M) x 64(P) tile: 4 accumulators, 4 back-to-back WMMAs per K-step.
template <int K>
__global__ void __launch_bounds__(256) k_wmma_gemm(const _Float16* __restrict__ Wt, // [Mpad][K]
                                                   const _Float16* __restrict__ X,  // [B][K][PIX]
                                                   int Mpad,
                                                   const float* __restrict__ res,   // [B][Mpad][PIX]|null
                                                   float* __restrict__ out32,
                                                   _Float16* __restrict__ out16) {
  constexpr int AS = K + 8;                       // padded LDS row stride (bank-conflict-free)
  __shared__ __align__(16) _Float16 As[16 * AS];
  int t = threadIdx.x;
  int lane = t & 31, wave = t >> 5;
  int g = lane >> 4, l16 = lane & 15;
  int bb = blockIdx.z;
  int m0 = blockIdx.y * 16;
  int p0 = blockIdx.x * 512 + wave * 64;

  // stage A tile (16 x K halfs) into LDS, coalesced 16B chunks
  constexpr int CH = K / 8;                       // 16B chunks per row
  for (int c = t; c < 16 * CH; c += 256) {
    int rr = c / CH, cc = (c % CH) * 8;
    *(uint4*)&As[rr * AS + cc] = *(const uint4*)&Wt[(size_t)(m0 + rr) * K + cc];
  }
  __syncthreads();

  v8f acc0 = {}, acc1 = {}, acc2 = {}, acc3 = {};
  const _Float16* Xb = X + (size_t)bb * K * PIX + p0 + l16;
  const _Float16* Ar = &As[l16 * AS];

  #pragma unroll
  for (int k0 = 0; k0 < K; k0 += 32) {
    if (k0 + 32 < K)
      __builtin_prefetch(Xb + (size_t)(k0 + 32) * PIX, 0, 1);  // global_prefetch_b8
    // A fragment (16x32 f16): VGPR r <- K = 16*(r/4) + 8*g + 2*(r%4) + {0,1}; pairs contiguous
    v16h a;
    #pragma unroll
    for (int r = 0; r < 8; ++r) {
      int kA = k0 + 16 * (r >> 2) + 8 * g + 2 * (r & 3);
      h2 pr = *(const h2*)&Ar[kA];
      a[2 * r]     = pr[0];
      a[2 * r + 1] = pr[1];
    }
    // 4 N-subtiles share the A fragment
    #pragma unroll
    for (int s = 0; s < 4; ++s) {
      // B fragment (32x16 f16): dword r = X[kB][p] | X[kB+1][p] << 16, kB = k0+16*(r/4)+8g+2*(r%4)
      // (explicit lo|hi<<16 merge so the backend can use global_load_d16 / _d16_hi pairs)
      v8u bu;
      #pragma unroll
      for (int r = 0; r < 8; ++r) {
        int kB = k0 + 16 * (r >> 2) + 8 * g + 2 * (r & 3);
        unsigned int lo =
            (unsigned int)__builtin_bit_cast(unsigned short, Xb[(size_t)kB * PIX + s * 16]);
        unsigned int hi =
            (unsigned int)__builtin_bit_cast(unsigned short, Xb[(size_t)(kB + 1) * PIX + s * 16]);
        bu[r] = lo | (hi << 16);
      }
      v16h bf = __builtin_bit_cast(v16h, bu);
      v8f& acc = (s == 0) ? acc0 : (s == 1) ? acc1 : (s == 2) ? acc2 : acc3;
      acc = __builtin_amdgcn_wmma_f32_16x16x32_f16(false, a, false, bf, (short)0, acc,
                                                   false, false);
    }
  }

  // C/D layout: element i -> M = m0 + 8*g + i, N = p0 + s*16 + lane%16
  #pragma unroll
  for (int s = 0; s < 4; ++s) {
    v8f& acc = (s == 0) ? acc0 : (s == 1) ? acc1 : (s == 2) ? acc2 : acc3;
    #pragma unroll
    for (int i = 0; i < 8; ++i) {
      int m = m0 + 8 * g + i;
      size_t off = ((size_t)bb * Mpad + m) * PIX + p0 + s * 16 + l16;
      float v = acc[i];
      if (res) v += res[off];
      if (out32) out32[off] = v;
      else out16[off] = (_Float16)v;
    }
  }
}

extern "C" void kernel_launch(void* const* d_in, const int* in_sizes, int n_in,
                              void* d_out, int out_size, void* d_ws, size_t ws_size,
                              hipStream_t stream) {
  const float* x         = (const float*)d_in[0];
  const float* w_local   = (const float*)d_in[1];
  const float* gamma1    = (const float*)d_in[2];
  const float* beta1     = (const float*)d_in[3];
  const float* w_qkv     = (const float*)d_in[4];
  const float* w_qkv_dw  = (const float*)d_in[5];
  const float* w_attnout = (const float*)d_in[6];
  const float* gamma2    = (const float*)d_in[7];
  const float* beta2     = (const float*)d_in[8];
  const float* w_in      = (const float*)d_in[9];
  const float* w_dw      = (const float*)d_in[10];
  const float* w_out     = (const float*)d_in[11];
  float* out = (float*)d_out;

  // ---- workspace carve-up ----
  char* ws = (char*)d_ws;
  size_t o = 0;
  float*    xmain = (float*)(ws + o);    o += (size_t)BATCH * DIM   * PIX * 4;  // 50.3 MB
  _Float16* f16a  = (_Float16*)(ws + o); o += (size_t)BATCH * DIM   * PIX * 2;  // 25.2 MB (xn/yn)
  _Float16* big   = (_Float16*)(ws + o); o += (size_t)BATCH * HID2P * PIX * 2;  // 134.2 MB (qkv_pre / y_pre)
  _Float16* qkvb  = (_Float16*)(ws + o); o += (size_t)BATCH * C3    * PIX * 2;  // 75.5 MB (qkv / g@512-stride)
  _Float16* aout  = (_Float16*)(ws + o); o += (size_t)BATCH * DIM   * PIX * 2;  // 25.2 MB
  float*    attnb = (float*)(ws + o);    o += (size_t)BATCH * HEADS * DH * DH * 4;
  _Float16* w_qkv16 = (_Float16*)(ws + o); o += (size_t)C3    * DIM  * 2;
  _Float16* w_ao16  = (_Float16*)(ws + o); o += (size_t)DIM   * DIM  * 2;
  _Float16* w_in16  = (_Float16*)(ws + o); o += (size_t)HID2P * DIM  * 2;  // 1024 rows (4 zero)
  _Float16* w_out16 = (_Float16*)(ws + o); o += (size_t)DIM   * HIDP * 2;  // 512 cols (2 zero)

  // ---- weights -> f16 (zero-padded) ----
  k_cvt_pad<<<(C3 * DIM + 255) / 256, 256, 0, stream>>>(w_qkv, w_qkv16, C3, DIM, C3, DIM);
  k_cvt_pad<<<(DIM * DIM + 255) / 256, 256, 0, stream>>>(w_attnout, w_ao16, DIM, DIM, DIM, DIM);
  k_cvt_pad<<<(HID2P * DIM + 255) / 256, 256, 0, stream>>>(w_in, w_in16, HID2, DIM, HID2P, DIM);
  k_cvt_pad<<<(DIM * HIDP + 255) / 256, 256, 0, stream>>>(w_out, w_out16, DIM, HID, DIM, HIDP);

  // ---- xn = LN(x) ----
  k_chan_ln<<<(BATCH * PIX) / 256, 256, 0, stream>>>(x, gamma1, beta1, f16a, DIM);

  // ---- qkv_pre = w_qkv @ xn  ([B][576][PIX] into big buffer) ----
  {
    dim3 g(PIX / 512, C3 / 16, BATCH);
    k_wmma_gemm<DIM><<<g, 256, 0, stream>>>(w_qkv16, f16a, C3, nullptr, nullptr, big);
  }
  // ---- qkv = dwconv3(qkv_pre) ----
  k_dwconv3_f16<<<(BATCH * C3 * PIX) / 256, 256, 0, stream>>>(big, w_qkv_dw, qkvb, C3);

  // ---- channel attention ----
  k_attn_scores<<<BATCH * HEADS, 576, 0, stream>>>(qkvb, attnb);
  k_attn_apply<<<dim3(PIX / 256, BATCH * HEADS), 256, 0, stream>>>(qkvb, attnb, aout);

  // ---- x_main = x + w_attn_out @ aout ----
  {
    dim3 g(PIX / 512, DIM / 16, BATCH);
    k_wmma_gemm<DIM><<<g, 256, 0, stream>>>(w_ao16, aout, DIM, x, xmain, nullptr);
  }

  // ---- yn = LN(x_main) ----
  k_chan_ln<<<(BATCH * PIX) / 256, 256, 0, stream>>>(xmain, gamma2, beta2, f16a, DIM);

  // ---- y_pre = w_in @ yn  ([B][1024][PIX], rows 1020..1023 are zero-weight rows) ----
  {
    dim3 g(PIX / 512, HID2P / 16, BATCH);
    k_wmma_gemm<DIM><<<g, 256, 0, stream>>>(w_in16, f16a, HID2P, nullptr, nullptr, big);
  }
  // ---- g = gelu(dw(y1)) * dw(y2) fused  -> [B][512-stride][PIX], 510 valid ch ----
  k_dwconv_geglu<<<(BATCH * HID * PIX) / 256, 256, 0, stream>>>(big, w_dw, qkvb);

  // ---- d_out = x_main + w_out @ g  (K=512, cols 510/511 of w_out16 are zero) ----
  {
    dim3 g(PIX / 512, DIM / 16, BATCH);
    k_wmma_gemm<HIDP><<<g, 256, 0, stream>>>(w_out16, qkvb, DIM, xmain, out, nullptr);
  }
  // ---- d_out += gelu(dwconv3(x, w_local)) ----
  k_local_ft_add<<<(BATCH * DIM * PIX) / 256, 256, 0, stream>>>(x, w_local, out);
}
